// RingNet_25314537243110
// MI455X (gfx1250) — matrix operand order repaired and checked by reading, not compiled, not run on hardware
//
#include <hip/hip_runtime.h>

typedef unsigned short u16;
typedef unsigned int   u32;

typedef __attribute__((ext_vector_type(16))) __bf16 v16bf;
typedef __attribute__((ext_vector_type(8)))  float  v8f;

#define HSZ   495
#define WSZ   436
#define NPIX  215820              // 495*436
#define NPAD  215872              // 3373*64  (GEMM1 N-tile padding)
#define PW    437                 // integral image width  (WSZ+1)
#define PH    496                 // integral image height (HSZ+1)
#define SPLANE (PW*PH)            // 216752 floats per channel
#define CIN   96
#define CCAT  1248
#define CMID  512
#define COUT  48
#define KT1   39                  // 1248 / 32
#define KT2   16                  // 512  / 32
#define INV_N (1.0f/215820.0f)
#define BN_EPS 1e-5f

__device__ __forceinline__ u16 f2bf(float f) {
  u32 u = __float_as_uint(f);
  u32 r = (u + 0x7FFFu + ((u >> 16) & 1u)) >> 16;   // round-to-nearest-even
  return (u16)r;
}
__device__ __forceinline__ float bf2f(u32 hs) { return __uint_as_float(hs << 16); }

// ---------------------------------------------------------------------------
// K1: per-(channel,row) prefix sum along W into padded integral image S.
// S[c] is PH x PW with zero top row / left column.
__global__ __launch_bounds__(256) void k_row_prefix(const float* __restrict__ x,
                                                    float* __restrict__ S) {
  int id = blockIdx.x * 256 + threadIdx.x;
  if (id >= CIN * PH) return;
  int c = id / PH, i = id % PH;
  float* Sr = S + (long)c * SPLANE + (long)i * PW;
  if (i == 0) {
    for (int w = 0; w < PW; ++w) Sr[w] = 0.0f;
    return;
  }
  const float* xr = x + (long)c * NPIX + (long)(i - 1) * WSZ;
  Sr[0] = 0.0f;
  float run = 0.0f;
  for (int w = 0; w < WSZ; ++w) { run += xr[w]; Sr[w + 1] = run; }
}

// K2: column prefix sum down each integral-image column.
__global__ __launch_bounds__(256) void k_col_prefix(float* __restrict__ S) {
  int id = blockIdx.x * 256 + threadIdx.x;
  if (id >= CIN * PW) return;
  int c = id / PW, j = id % PW;
  float* p = S + (long)c * SPLANE + j;
  float acc = p[0];
  for (int i = 1; i < PH; ++i) { acc += p[(long)i * PW]; p[(long)i * PW] = acc; }
}

// pack 32 floats -> 32 bf16 and store as 4x uint4 (64B contiguous)
__device__ __forceinline__ void store32(u16* __restrict__ d, const float* v) {
#pragma unroll
  for (int q = 0; q < 4; ++q) {
    uint4 pk;
    pk.x = (u32)f2bf(v[q*8+0]) | ((u32)f2bf(v[q*8+1]) << 16);
    pk.y = (u32)f2bf(v[q*8+2]) | ((u32)f2bf(v[q*8+3]) << 16);
    pk.z = (u32)f2bf(v[q*8+4]) | ((u32)f2bf(v[q*8+5]) << 16);
    pk.w = (u32)f2bf(v[q*8+6]) | ((u32)f2bf(v[q*8+7]) << 16);
    *(uint4*)(d + q * 8) = pk;
  }
}

// K3: build pixel-major bf16 feature matrix Ht[n][1248]:
//     group 0 = identity, group g>=1 = ring(k=2g+1) = box(g) - box(g-1)
__global__ __launch_bounds__(256) void k_build_features(const float* __restrict__ x,
                                                        const float* __restrict__ S,
                                                        u16* __restrict__ Ht) {
  int p = blockIdx.x * 256 + threadIdx.x;
  if (p >= NPIX) return;
  int h = p / WSZ, w = p - h * WSZ;
  u16* dst = Ht + (long)p * CCAT;

  // identity branch
#pragma unroll
  for (int cb = 0; cb < 3; ++cb) {
    float v[32];
#pragma unroll
    for (int j = 0; j < 32; ++j) v[j] = x[(long)(cb * 32 + j) * NPIX + p];
    store32(dst + cb * 32, v);
  }

  // ring branches via integral image
  for (int g = 1; g < 13; ++g) {
    int r0o = max(h - g, 0),       r1o = min(h + g, HSZ - 1) + 1;
    int c0o = max(w - g, 0),       c1o = min(w + g, WSZ - 1) + 1;
    int gi = g - 1;
    int r0i = max(h - gi, 0),      r1i = min(h + gi, HSZ - 1) + 1;
    int c0i = max(w - gi, 0),      c1i = min(w + gi, WSZ - 1) + 1;
    int o00 = r0o * PW + c0o, o01 = r0o * PW + c1o;
    int o10 = r1o * PW + c0o, o11 = r1o * PW + c1o;
    int i00 = r0i * PW + c0i, i01 = r0i * PW + c1i;
    int i10 = r1i * PW + c0i, i11 = r1i * PW + c1i;
#pragma unroll
    for (int cb = 0; cb < 3; ++cb) {
      float v[32];
#pragma unroll
      for (int j = 0; j < 32; ++j) {
        const float* Sc = S + (long)(cb * 32 + j) * SPLANE;
        float outer = Sc[o11] - Sc[o01] - Sc[o10] + Sc[o00];
        float inner = Sc[i11] - Sc[i01] - Sc[i10] + Sc[i00];
        v[j] = outer - inner;
      }
      store32(dst + g * 96 + cb * 32, v);
    }
  }
}

// K4a: per-block partial sums/sumsq over channel pairs of Ht (deterministic tree)
__global__ __launch_bounds__(640) void k_stats1_partial(const u16* __restrict__ Ht,
                                                        float* __restrict__ part) {
  int t = threadIdx.x, b = blockIdx.x;
  if (t >= 624) return;
  const int chunk = (NPIX + 511) / 512;
  int p0 = b * chunk, p1 = min(NPIX, p0 + chunk);
  float s0 = 0, q0 = 0, s1 = 0, q1 = 0;
  for (int p = p0; p < p1; ++p) {
    u32 u = *(const u32*)(Ht + (long)p * CCAT + 2 * t);
    float a = bf2f(u & 0xffffu), c = bf2f(u >> 16);
    s0 += a; q0 += a * a; s1 += c; q1 += c * c;
  }
  float* dp = part + (long)b * (2 * CCAT) + 4 * t;
  dp[0] = s0; dp[1] = q0; dp[2] = s1; dp[3] = q1;
}

// K: generic fixed-order reduction over 512 partial blocks
__global__ __launch_bounds__(256) void k_stats_reduce(const float* __restrict__ part,
                                                      float* __restrict__ out, int nvals) {
  int i = blockIdx.x * 256 + threadIdx.x;
  if (i >= nvals) return;
  float s = 0.0f;
  for (int b = 0; b < 512; ++b) s += part[(long)b * nvals + i];
  out[i] = s;
}

// K7a: stats over mid tensor (512 channels, pixel-major)
__global__ __launch_bounds__(256) void k_stats2_partial(const u16* __restrict__ midT,
                                                        float* __restrict__ part) {
  int t = threadIdx.x, b = blockIdx.x;
  const int chunk = (NPIX + 511) / 512;
  int p0 = b * chunk, p1 = min(NPIX, p0 + chunk);
  float s0 = 0, q0 = 0, s1 = 0, q1 = 0;
  for (int p = p0; p < p1; ++p) {
    u32 u = *(const u32*)(midT + (long)p * CMID + 2 * t);
    float a = bf2f(u & 0xffffu), c = bf2f(u >> 16);
    s0 += a; q0 += a * a; s1 += c; q1 += c * c;
  }
  float* dp = part + (long)b * (2 * CMID) + 4 * t;
  dp[0] = s0; dp[1] = q0; dp[2] = s1; dp[3] = q1;
}

// Fold BN1 into w1: write A-fragment-swizzled bf16 weights + folded bias.
// Swizzle: K-tile kt, r=k%32 -> half=r>>4, slot=r&15
__global__ __launch_bounds__(256) void k_fold1(const float* __restrict__ w1,
                                               const float* __restrict__ b1,
                                               const float* __restrict__ g1,
                                               const float* __restrict__ be1,
                                               const float* __restrict__ stats,
                                               u16* __restrict__ w1s,
                                               float* __restrict__ b1f) {
  int m = blockIdx.x, t = threadIdx.x;
  __shared__ float red[256];
  float partial = 0.0f;
  for (int k = t; k < CCAT; k += 256) {
    float mean  = stats[2 * k] * INV_N;
    float var   = stats[2 * k + 1] * INV_N - mean * mean;
    float scale = g1[k] * rsqrtf(var + BN_EPS);
    float shift = be1[k] - mean * scale;
    float wv = w1[(long)m * CCAT + k];
    int kt = k >> 5, r = k & 31;
    int hf = r >> 4, s = r & 15;
    w1s[(((long)m * KT1 + kt) * 2 + hf) * 16 + s] = f2bf(wv * scale);
    partial += wv * shift;
  }
  red[t] = partial; __syncthreads();
  for (int o = 128; o > 0; o >>= 1) { if (t < o) red[t] += red[t + o]; __syncthreads(); }
  if (t == 0) b1f[m] = b1[m] + red[0];
}

// Fold BN2 into w2 (rows 48..63 zero-padded), same swizzle, K=512.
__global__ __launch_bounds__(256) void k_fold2(const float* __restrict__ w2,
                                               const float* __restrict__ b2,
                                               const float* __restrict__ g2,
                                               const float* __restrict__ be2,
                                               const float* __restrict__ stats,
                                               u16* __restrict__ w2s,
                                               float* __restrict__ b2f) {
  int m = blockIdx.x, t = threadIdx.x;
  __shared__ float red[256];
  float partial = 0.0f;
  for (int k = t; k < CMID; k += 256) {
    float mean  = stats[2 * k] * INV_N;
    float var   = stats[2 * k + 1] * INV_N - mean * mean;
    float scale = g2[k] * rsqrtf(var + BN_EPS);
    float shift = be2[k] - mean * scale;
    float wv = (m < COUT) ? w2[(long)m * CMID + k] : 0.0f;
    int kt = k >> 5, r = k & 31;
    int hf = r >> 4, s = r & 15;
    w2s[(((long)m * KT2 + kt) * 2 + hf) * 16 + s] = f2bf(wv * scale);
    partial += wv * shift;
  }
  red[t] = partial; __syncthreads();
  for (int o = 128; o > 0; o >>= 1) { if (t < o) red[t] += red[t + o]; __syncthreads(); }
  if (t == 0) b2f[m] = ((m < COUT) ? b2[m] : 0.0f) + red[0];
}

// GEMM1: mid = relu(w1f @ H + b1f).  M=512, K=1248, N=NPAD.
// Block tile 256(M) x 64(N); 8 waves as 4x2, each wave 64x32 = 4x2 WMMA tiles.
__global__ __launch_bounds__(256) void k_gemm1_wmma(const u16* __restrict__ w1s,
                                                    const float* __restrict__ b1f,
                                                    const u16* __restrict__ Ht,
                                                    u16* __restrict__ midT) {
  const int tid = threadIdx.x;
  const int wv = tid >> 5, lane = tid & 31;
  const int lr = lane & 15, half = lane >> 4;
  const int mBase = blockIdx.x * 256 + (wv >> 1) * 64;
  const long nBase = (long)blockIdx.y * 64 + (wv & 1) * 32;

  v8f acc[4][2];
#pragma unroll
  for (int i = 0; i < 4; ++i)
#pragma unroll
    for (int j = 0; j < 2; ++j)
#pragma unroll
      for (int r = 0; r < 8; ++r) acc[i][j][r] = 0.0f;

  for (int kt = 0; kt < KT1; ++kt) {
    v16bf a[4], b[2];
#pragma unroll
    for (int i = 0; i < 4; ++i) {
      int row = mBase + i * 16 + lr;
      a[i] = *(const v16bf*)(w1s + (((long)row * KT1 + kt) * 2 + half) * 16);
    }
#pragma unroll
    for (int j = 0; j < 2; ++j) {
      long col = nBase + j * 16 + lr;
      b[j] = *(const v16bf*)(Ht + col * CCAT + kt * 32 + half * 16);
      if (kt + 1 < KT1)
        __builtin_prefetch((const void*)(Ht + col * CCAT + (kt + 1) * 32 + half * 16), 0, 1);
    }
#pragma unroll
    for (int i = 0; i < 4; ++i)
#pragma unroll
      for (int j = 0; j < 2; ++j)
        acc[i][j] = __builtin_amdgcn_wmma_f32_16x16x32_bf16(
            false, a[i], false, b[j], (short)0, acc[i][j], false, false);
  }

  // epilogue: bias + relu + bf16, pixel-major store (8 contiguous channels/lane)
#pragma unroll
  for (int i = 0; i < 4; ++i) {
    int mloc = mBase + i * 16 + half * 8;
    float bias[8];
#pragma unroll
    for (int r = 0; r < 8; ++r) bias[r] = b1f[mloc + r];
#pragma unroll
    for (int j = 0; j < 2; ++j) {
      long n = nBase + j * 16 + lr;
      float v[8];
#pragma unroll
      for (int r = 0; r < 8; ++r) v[r] = fmaxf(acc[i][j][r] + bias[r], 0.0f);
      uint4 pk;
      pk.x = (u32)f2bf(v[0]) | ((u32)f2bf(v[1]) << 16);
      pk.y = (u32)f2bf(v[2]) | ((u32)f2bf(v[3]) << 16);
      pk.z = (u32)f2bf(v[4]) | ((u32)f2bf(v[5]) << 16);
      pk.w = (u32)f2bf(v[6]) | ((u32)f2bf(v[7]) << 16);
      *(uint4*)(midT + n * CMID + mloc) = pk;
    }
  }
}

// GEMM2: out = w2f @ mid + b2f.  M=64(pad of 48), K=512, N tile 32 per block.
__global__ __launch_bounds__(256) void k_gemm2_wmma(const u16* __restrict__ w2s,
                                                    const float* __restrict__ b2f,
                                                    const u16* __restrict__ midT,
                                                    float* __restrict__ out) {
  const int tid = threadIdx.x;
  const int wv = tid >> 5, lane = tid & 31;
  const int lr = lane & 15, half = lane >> 4;
  const int i = wv >> 1;                       // M sub-tile (0..3)
  const long nBase = (long)blockIdx.x * 32 + (wv & 1) * 16;

  v8f acc;
#pragma unroll
  for (int r = 0; r < 8; ++r) acc[r] = 0.0f;

  for (int kt = 0; kt < KT2; ++kt) {
    int row = i * 16 + lr;
    v16bf a = *(const v16bf*)(w2s + (((long)row * KT2 + kt) * 2 + half) * 16);
    v16bf b = *(const v16bf*)(midT + (nBase + lr) * CMID + kt * 32 + half * 16);
    acc = __builtin_amdgcn_wmma_f32_16x16x32_bf16(
        false, a, false, b, (short)0, acc, false, false);
  }

  long n = nBase + lr;
  if (n < NPIX) {
    int mloc = i * 16 + half * 8;
#pragma unroll
    for (int r = 0; r < 8; ++r) {
      int m = mloc + r;
      if (m < COUT) out[(long)m * NPIX + n] = acc[r] + b2f[m];
    }
  }
}

// ---------------------------------------------------------------------------
extern "C" void kernel_launch(void* const* d_in, const int* in_sizes, int n_in,
                              void* d_out, int out_size, void* d_ws, size_t ws_size,
                              hipStream_t stream) {
  (void)in_sizes; (void)n_in; (void)out_size; (void)ws_size;
  const float* x   = (const float*)d_in[0];
  const float* g1  = (const float*)d_in[1];
  const float* be1 = (const float*)d_in[2];
  const float* w1  = (const float*)d_in[3];
  const float* b1  = (const float*)d_in[4];
  const float* g2  = (const float*)d_in[5];
  const float* be2 = (const float*)d_in[6];
  const float* w2  = (const float*)d_in[7];
  const float* b2  = (const float*)d_in[8];
  float* out = (float*)d_out;

  // workspace carve (256B aligned slices)
  char* p = (char*)d_ws;
  auto carve = [&](size_t bytes) -> void* {
    void* r = (void*)p;
    p += (bytes + 255) & ~(size_t)255;
    return r;
  };
  float* S      = (float*)carve((size_t)CIN * SPLANE * 4);            // 83.2 MB
  u16*   Ht     = (u16*)  carve((size_t)NPAD * CCAT * 2);             // 538.8 MB
  u16*   midT   = (u16*)  carve((size_t)NPAD * CMID * 2);             // 221.1 MB
  u16*   w1s    = (u16*)  carve((size_t)CMID * CCAT * 2);
  float* b1f    = (float*)carve((size_t)CMID * 4);
  u16*   w2s    = (u16*)  carve((size_t)64 * CMID * 2);
  float* b2f    = (float*)carve((size_t)64 * 4);
  float* stats1 = (float*)carve((size_t)2 * CCAT * 4);
  float* stats2 = (float*)carve((size_t)2 * CMID * 4);
  float* part1  = (float*)carve((size_t)512 * 2 * CCAT * 4);          // 5.1 MB
  float* part2  = (float*)carve((size_t)512 * 2 * CMID * 4);          // 2.1 MB

  // 1-2: integral images
  k_row_prefix<<<(CIN * PH + 255) / 256, 256, 0, stream>>>(x, S);
  k_col_prefix<<<(CIN * PW + 255) / 256, 256, 0, stream>>>(S);
  // 3: ring features (bf16, pixel-major)
  k_build_features<<<(NPIX + 255) / 256, 256, 0, stream>>>(x, S, Ht);
  // 4: BN1 stats (deterministic two-stage)
  k_stats1_partial<<<512, 640, 0, stream>>>(Ht, part1);
  k_stats_reduce<<<(2 * CCAT + 255) / 256, 256, 0, stream>>>(part1, stats1, 2 * CCAT);
  // 5: fold BN1 into w1 (bf16, A-fragment swizzled)
  k_fold1<<<CMID, 256, 0, stream>>>(w1, b1, g1, be1, stats1, w1s, b1f);
  // 6: big WMMA GEMM + bias + relu -> mid (bf16, pixel-major)
  k_gemm1_wmma<<<dim3(CMID / 256, NPAD / 64), 256, 0, stream>>>(w1s, b1f, Ht, midT);
  // 7: BN2 stats
  k_stats2_partial<<<512, 256, 0, stream>>>(midT, part2);
  k_stats_reduce<<<(2 * CMID + 255) / 256, 256, 0, stream>>>(part2, stats2, 2 * CMID);
  // 8: fold BN2 into w2
  k_fold2<<<64, 256, 0, stream>>>(w2, b2, g2, be2, stats2, w2s, b2f);
  // 9: output WMMA GEMM -> d_out (fp32)
  k_gemm2_wmma<<<(NPIX + 31) / 32, 256, 0, stream>>>(w2s, b2f, midT, out);
}
